// NeighborhoodAttention2D_41437844471830
// MI455X (gfx1250) — matrix operand order repaired and checked by reading, not compiled, not run on hardware
//
#include <hip/hip_runtime.h>
#include <hip/hip_bf16.h>
#include <stdint.h>

// ---------------- constants ----------------
#define KSZ     7
#define HEADS   16
#define DIM     512
#define DHEAD   32
#define HH      56
#define WW      56
#define HW      3136       // 56*56
#define CHW     1605632    // 512*3136
#define MTOT    25088      // 8*3136
#define NQKV    1536       // 3*512
#define KSTEPS  16         // 512 / 32

typedef __attribute__((ext_vector_type(16))) __bf16 v16bf;
typedef __attribute__((ext_vector_type(8)))  float  v8f;

union BF8U4 { uint4 u; __bf16 h[8]; };
union FRAG  { uint4 u[2]; v16bf v; };

// ---- CDNA5 async global->LDS copy (ASYNCcnt-tracked, no VGPR staging) ----
__device__ __forceinline__ void async_copy_b128(uint32_t lds_addr, const void* gaddr) {
  asm volatile("global_load_async_to_lds_b128 %0, %1, off"
               :: "v"(lds_addr), "v"((uint64_t)(uintptr_t)gaddr)
               : "memory");
}
__device__ __forceinline__ void wait_async0() {
  asm volatile("s_wait_asynccnt 0" ::: "memory");
}
__device__ __forceinline__ uint32_t lds_addr_of(const void* p) {
  // generic LDS address: low 32 bits are the in-LDS offset (aperture in [63:32])
  return (uint32_t)(uintptr_t)p;
}

// =====================================================================
// One-time weight conversion fp32 -> bf16 (weights: 4 MB fp32 total, L2-resident)
// =====================================================================
__global__ __launch_bounds__(256) void cvt_w_kernel(
    const float* __restrict__ qw, const float* __restrict__ pw,
    __bf16* __restrict__ qbf, __bf16* __restrict__ pbf) {
  const int i = (blockIdx.x * 256 + threadIdx.x) * 4;   // 1024 blocks: 1048576 elems
  if (i < NQKV * DIM) {
    const float4 v = *(const float4*)(qw + i);
    qbf[i + 0] = (__bf16)v.x; qbf[i + 1] = (__bf16)v.y;
    qbf[i + 2] = (__bf16)v.z; qbf[i + 3] = (__bf16)v.w;
  } else {
    const int j = i - NQKV * DIM;
    const float4 v = *(const float4*)(pw + j);
    pbf[j + 0] = (__bf16)v.x; pbf[j + 1] = (__bf16)v.y;
    pbf[j + 2] = (__bf16)v.z; pbf[j + 3] = (__bf16)v.w;
  }
}

// =====================================================================
// GEMM 1: qkv[M,1536](bf16) = x_nchw[M,512](f32) @ qkv_w^T
// 128x128 block tile, K-step 32, 8 waves (4Mx2N) -> 8 WMMA / wave / step.
// Ping-pong LDS: tile i+1 copies (async B, staged A) overlap WMMA of tile i.
// =====================================================================
__global__ __launch_bounds__(256) void qkv_gemm_kernel(
    const float* __restrict__ x, const __bf16* __restrict__ wq_bf,
    __bf16* __restrict__ qkv_out) {
  __shared__ __bf16 a_tile[2][128][40];   // [buf][m][k], 80B rows (16B aligned)
  __shared__ __bf16 b_tile[2][128][40];   // [buf][n][k]

  const int tid  = threadIdx.x;
  const int m0   = blockIdx.x * 128;
  const int n0   = blockIdx.y * 128;
  const int lane = tid & 31;
  const int wv   = tid >> 5;
  const int wm   = wv >> 1;            // 0..3 : 32 rows each
  const int wn   = wv & 1;             // 0..1 : 64 cols each
  const int lrow = lane & 15;
  const int lhi  = lane >> 4;

  // A loader: row mm, k-half kk0 (128x32 tile, 16 elems/thread)
  const int mm  = tid & 127;
  const int kk0 = (tid >> 7) << 4;     // 0 or 16
  const int m_t = m0 + mm;
  const int img = m_t / HW;            // 128-row tiles may straddle images
  const int rem = m_t % HW;

  // B async loader: row bln, 32B (16 bf16) at kb
  const int bln = tid >> 1;
  const int kb  = (tid & 1) * 16;

  auto load_b = [&](int k0, int buf) {
    const __bf16* src = wq_bf + (size_t)(n0 + bln) * DIM + k0 + kb;
    async_copy_b128(lds_addr_of(&b_tile[buf][bln][kb]),     src);
    async_copy_b128(lds_addr_of(&b_tile[buf][bln][kb + 8]), src + 8);
  };
  auto load_a = [&](int k0, int buf) {
    const float* ap = x + (size_t)img * CHW + (size_t)(k0 + kk0) * HW + rem;
    BF8U4 p0, p1;
    #pragma unroll
    for (int q = 0; q < 8; ++q) p0.h[q] = (__bf16)ap[(size_t)q * HW];
    #pragma unroll
    for (int q = 0; q < 8; ++q) p1.h[q] = (__bf16)ap[(size_t)(q + 8) * HW];
    *(uint4*)&a_tile[buf][mm][kk0]     = p0.u;
    *(uint4*)&a_tile[buf][mm][kk0 + 8] = p1.u;
  };

  v8f acc[2][4];
  #pragma unroll
  for (int a = 0; a < 2; ++a)
    #pragma unroll
    for (int b = 0; b < 4; ++b) acc[a][b] = (v8f){};

  // prologue: fill buffer 0
  load_b(0, 0);
  load_a(0, 0);
  wait_async0();
  __syncthreads();

  for (int it = 0; it < KSTEPS; ++it) {
    const int cur = it & 1;
    // prefetch tile it+1 into the other buffer (overlaps with WMMAs below)
    if (it + 1 < KSTEPS) {
      load_b((it + 1) * 32, cur ^ 1);
      load_a((it + 1) * 32, cur ^ 1);
    }

    FRAG af[2];
    #pragma unroll
    for (int a = 0; a < 2; ++a) {
      const int row = wm * 32 + a * 16 + lrow;
      af[a].u[0] = *(const uint4*)&a_tile[cur][row][lhi * 16];
      af[a].u[1] = *(const uint4*)&a_tile[cur][row][lhi * 16 + 8];
    }
    FRAG bfr[4];
    #pragma unroll
    for (int b = 0; b < 4; ++b) {
      const int col = wn * 64 + b * 16 + lrow;
      bfr[b].u[0] = *(const uint4*)&b_tile[cur][col][lhi * 16];
      bfr[b].u[1] = *(const uint4*)&b_tile[cur][col][lhi * 16 + 8];
    }
    #pragma unroll
    for (int a = 0; a < 2; ++a)
      #pragma unroll
      for (int b = 0; b < 4; ++b)
        acc[a][b] = __builtin_amdgcn_wmma_f32_16x16x32_bf16(
            false, af[a].v, false, bfr[b].v, (short)0, acc[a][b], false, false);

    wait_async0();      // tile it+1 copies done (issued before the WMMAs)
    __syncthreads();    // all waves' stores for it+1 visible; reads of cur done
  }

  // D layout: lane -> N = lrow, VGPR r -> M = r + 8*lhi
  #pragma unroll
  for (int a = 0; a < 2; ++a) {
    const int rowb = m0 + wm * 32 + a * 16 + 8 * lhi;
    #pragma unroll
    for (int b = 0; b < 4; ++b) {
      const int col = n0 + wn * 64 + b * 16 + lrow;
      #pragma unroll
      for (int r = 0; r < 8; ++r)
        qkv_out[(size_t)(rowb + r) * NQKV + col] = (__bf16)acc[a][b][r];
    }
  }
}

// =====================================================================
// Neighborhood attention, flash-style online softmax (qkv bf16 is L2-resident)
// =====================================================================
__global__ __launch_bounds__(256) void nat_attn_kernel(
    const __bf16* __restrict__ qkv, const float* __restrict__ rpb,
    __bf16* __restrict__ attn_out) {
  const int idx = blockIdx.x * 256 + threadIdx.x;
  const int j = idx % WW;
  const int i = (idx / WW) % HH;
  const int h = (idx / HW) % HEADS;
  const int b = idx / (HW * HEADS);
  const int m = (b * HH + i) * WW + j;

  const float scale = 0.17677669529663687f; // 32^-0.5

  float q[32];
  {
    const uint4* qp = (const uint4*)(qkv + (size_t)m * NQKV + h * DHEAD);
    #pragma unroll
    for (int t = 0; t < 4; ++t) {
      const uint4 u = qp[t];
      const unsigned ua[4] = {u.x, u.y, u.z, u.w};
      #pragma unroll
      for (int e = 0; e < 4; ++e) {
        q[t * 8 + 2 * e]     = __uint_as_float(ua[e] << 16) * scale;
        q[t * 8 + 2 * e + 1] = __uint_as_float(ua[e] & 0xffff0000u) * scale;
      }
    }
  }

  int si = i - 3; si = si < 0 ? 0 : (si > HH - KSZ ? HH - KSZ : si);
  int sj = j - 3; sj = sj < 0 ? 0 : (sj > WW - KSZ ? WW - KSZ : sj);

  float mrun = -3.0e38f, srun = 0.f;
  float out[32];
  #pragma unroll
  for (int t = 0; t < 32; ++t) out[t] = 0.f;

  for (int a = 0; a < KSZ; ++a) {
    const int ki = si + a;
    const float* biasrow = rpb + h * 169 + (ki - i + 6) * 13 + (sj - j + 6);
    for (int c = 0; c < KSZ; ++c) {
      const int kj = sj + c;
      const size_t km = (size_t)(b * HH + ki) * WW + kj;
      const uint4* kp = (const uint4*)(qkv + km * NQKV + DIM + h * DHEAD);
      const uint4* vp = (const uint4*)(qkv + km * NQKV + 2 * DIM + h * DHEAD);
      float logit = biasrow[c];
      float vf[32];
      #pragma unroll
      for (int t = 0; t < 4; ++t) {
        const uint4 uk = kp[t];
        const uint4 uv = vp[t];
        const unsigned ka[4] = {uk.x, uk.y, uk.z, uk.w};
        const unsigned va[4] = {uv.x, uv.y, uv.z, uv.w};
        #pragma unroll
        for (int e = 0; e < 4; ++e) {
          logit = fmaf(q[t * 8 + 2 * e],     __uint_as_float(ka[e] << 16),         logit);
          logit = fmaf(q[t * 8 + 2 * e + 1], __uint_as_float(ka[e] & 0xffff0000u), logit);
          vf[t * 8 + 2 * e]     = __uint_as_float(va[e] << 16);
          vf[t * 8 + 2 * e + 1] = __uint_as_float(va[e] & 0xffff0000u);
        }
      }
      const float newmax = fmaxf(mrun, logit);
      const float corr   = __expf(mrun - newmax);
      const float p      = __expf(logit - newmax);
      srun = srun * corr + p;
      #pragma unroll
      for (int t = 0; t < 32; ++t) out[t] = out[t] * corr + p * vf[t];
      mrun = newmax;
    }
  }

  const float inv = 1.f / srun;
  __bf16* op = attn_out + (size_t)m * DIM + h * DHEAD;
  #pragma unroll
  for (int t = 0; t < 4; ++t) {
    BF8U4 pk;
    #pragma unroll
    for (int e = 0; e < 8; ++e) pk.h[e] = (__bf16)(out[t * 8 + e] * inv);
    *(uint4*)(op + t * 8) = pk.u;
  }
}

// =====================================================================
// GEMM 2: y_nchw[M,512](f32) = attn[M,512](bf16) @ proj_w^T + proj_b
// Both A and B tiles async global->LDS; ping-pong double buffering.
// =====================================================================
__global__ __launch_bounds__(256) void proj_gemm_kernel(
    const __bf16* __restrict__ a_in, const __bf16* __restrict__ wp_bf,
    const float* __restrict__ proj_b, float* __restrict__ y) {
  __shared__ __bf16 a_tile[2][128][40];
  __shared__ __bf16 b_tile[2][128][40];

  const int tid  = threadIdx.x;
  const int m0   = blockIdx.x * 128;
  const int n0   = blockIdx.y * 128;
  const int lane = tid & 31;
  const int wv   = tid >> 5;
  const int wm   = wv >> 1;
  const int wn   = wv & 1;
  const int lrow = lane & 15;
  const int lhi  = lane >> 4;

  const int rln = tid >> 1;            // loader row (m or n)
  const int kb  = (tid & 1) * 16;

  auto load_ab = [&](int k0, int buf) {
    const __bf16* asrc = a_in + (size_t)(m0 + rln) * DIM + k0 + kb;
    async_copy_b128(lds_addr_of(&a_tile[buf][rln][kb]),     asrc);
    async_copy_b128(lds_addr_of(&a_tile[buf][rln][kb + 8]), asrc + 8);
    const __bf16* bsrc = wp_bf + (size_t)(n0 + rln) * DIM + k0 + kb;
    async_copy_b128(lds_addr_of(&b_tile[buf][rln][kb]),     bsrc);
    async_copy_b128(lds_addr_of(&b_tile[buf][rln][kb + 8]), bsrc + 8);
  };

  v8f acc[2][4];
  #pragma unroll
  for (int a = 0; a < 2; ++a)
    #pragma unroll
    for (int b = 0; b < 4; ++b) acc[a][b] = (v8f){};

  load_ab(0, 0);
  wait_async0();
  __syncthreads();

  for (int it = 0; it < KSTEPS; ++it) {
    const int cur = it & 1;
    if (it + 1 < KSTEPS) load_ab((it + 1) * 32, cur ^ 1);

    FRAG af[2];
    #pragma unroll
    for (int a = 0; a < 2; ++a) {
      const int row = wm * 32 + a * 16 + lrow;
      af[a].u[0] = *(const uint4*)&a_tile[cur][row][lhi * 16];
      af[a].u[1] = *(const uint4*)&a_tile[cur][row][lhi * 16 + 8];
    }
    FRAG bfr[4];
    #pragma unroll
    for (int b = 0; b < 4; ++b) {
      const int col = wn * 64 + b * 16 + lrow;
      bfr[b].u[0] = *(const uint4*)&b_tile[cur][col][lhi * 16];
      bfr[b].u[1] = *(const uint4*)&b_tile[cur][col][lhi * 16 + 8];
    }
    #pragma unroll
    for (int a = 0; a < 2; ++a)
      #pragma unroll
      for (int b = 0; b < 4; ++b)
        acc[a][b] = __builtin_amdgcn_wmma_f32_16x16x32_bf16(
            false, af[a].v, false, bfr[b].v, (short)0, acc[a][b], false, false);

    wait_async0();
    __syncthreads();
  }

  // scatter to NCHW fp32 + bias (per-element img/rem: tiles straddle images)
  #pragma unroll
  for (int b = 0; b < 4; ++b) {
    const int col = n0 + wn * 64 + b * 16 + lrow;
    const float bias = proj_b[col];
    #pragma unroll
    for (int a = 0; a < 2; ++a) {
      const int rowb = m0 + wm * 32 + a * 16 + 8 * lhi;
      #pragma unroll
      for (int r = 0; r < 8; ++r) {
        const int mrow = rowb + r;
        const int img  = mrow / HW;
        const int rem  = mrow % HW;
        y[(size_t)img * CHW + (size_t)col * HW + rem] = acc[a][b][r] + bias;
      }
    }
  }
}

// =====================================================================
extern "C" void kernel_launch(void* const* d_in, const int* in_sizes, int n_in,
                              void* d_out, int out_size, void* d_ws, size_t ws_size,
                              hipStream_t stream) {
  const float* x      = (const float*)d_in[0];
  const float* qkv_w  = (const float*)d_in[1];
  const float* rpb    = (const float*)d_in[2];
  const float* proj_w = (const float*)d_in[3];
  const float* proj_b = (const float*)d_in[4];
  float* y = (float*)d_out;

  // workspace layout (bytes):
  //   qkv bf16    [25088 x 1536] = 77,070,336
  //   attn bf16   [25088 x  512] = 25,690,112
  //   qkv_w bf16  [1536 x 512]   =  1,572,864
  //   proj_w bf16 [ 512 x 512]   =    524,288   (total 100 MB)
  char* ws = (char*)d_ws;
  __bf16* qkv_ws  = (__bf16*)(ws);
  __bf16* attn_ws = (__bf16*)(ws + 77070336u);
  __bf16* qw_bf   = (__bf16*)(ws + 102760448u);
  __bf16* pw_bf   = (__bf16*)(ws + 104333312u);

  dim3 blk(256);
  cvt_w_kernel<<<dim3(1024), blk, 0, stream>>>(qkv_w, proj_w, qw_bf, pw_bf);
  qkv_gemm_kernel<<<dim3(MTOT / 128, NQKV / 128), blk, 0, stream>>>(x, qw_bf, qkv_ws);
  nat_attn_kernel<<<dim3((MTOT * HEADS) / 256), blk, 0, stream>>>(qkv_ws, rpb, attn_ws);
  proj_gemm_kernel<<<dim3(MTOT / 128, DIM / 128), blk, 0, stream>>>(attn_ws, pw_bf, proj_b, y);
}
